// SNN_60524679135295
// MI455X (gfx1250) — matrix operand order repaired and checked by reading, not compiled, not run on hardware
//
#include <hip/hip_runtime.h>
#include <cstdint>
#include <cstddef>

// ---------------------------------------------------------------------------
// SNN forward (snntorch Leaky, reset=subtract) on gfx1250.
//   B=2048, T=150, D_IN=20, H=256, OUT=200, beta=0.9, thr=1, warmup=50.
// Design (MI455X):
//  * f16 WMMA 16x16x32 w/ f32 accum. Spikes are exactly {0,1} in f16.
//  * Weights pre-packed into WMMA B-fragment order in d_ws (L2-resident,
//    ~504KB vs 192MB L2). Fragment loads are asm-volatile global_load_b128
//    batches of 4 K-fragments (one s_wait_loadcnt per 4 WMMAs) so LICM
//    cannot hoist the loop-invariant weight loads (which previously caused
//    >256-VGPR pressure and scratch spills).
//  * Spike activations staged row-major f16 in LDS -> A fragment = two
//    aligned ds_load_b128 per lane (ISA 7.12.2 A layout).
//  * Membranes m1..m4, mo and the softmax accumulator live in VGPRs
//    (C/D fragment layout) across the whole 150-step recurrence.
//  * One block = 16 batch rows for all T; 8 waves split N; grid = 128.
// ---------------------------------------------------------------------------

#define BATCH   2048
#define TSTEPS  150
#define DIN     20
#define HID     256
#define NOUT    200
#define BETAF   0.9f
#define THRF    1.0f
#define WARMUP  50

typedef __attribute__((ext_vector_type(16))) _Float16 v16h;
typedef __attribute__((ext_vector_type(8)))  _Float16 v8h;
typedef __attribute__((ext_vector_type(8)))  float    v8f;

union Frag16 { v16h v; v8h h[2]; };

__device__ __forceinline__ v8f wmma_f16f32(v16h a, v16h b, v8f c) {
  // D = A(16x32 f16) * B(32x16 f16) + C(16x16 f32)
  return __builtin_amdgcn_wmma_f32_16x16x32_f16(
      /*neg_a=*/false, a, /*neg_b=*/false, b,
      /*c_mod=*/(short)0, c, /*reuse_a=*/false, /*reuse_b=*/false);
}

__device__ __forceinline__ v16h mkfrag(v8h lo, v8h hi) {
  Frag16 f; f.h[0] = lo; f.h[1] = hi; return f.v;
}

// One B fragment (32B per lane): two global_load_b128 + wait. asm volatile so
// it cannot be hoisted out of the t-loop.
__device__ __forceinline__ v16h load_b1(const _Float16* p) {
  v8h lo, hi;
  asm volatile(
      "global_load_b128 %0, %2, off\n\t"
      "global_load_b128 %1, %2, off offset:16\n\t"
      "s_wait_loadcnt 0x0"
      : "=v"(lo), "=v"(hi)
      : "v"(p));
  return mkfrag(lo, hi);
}

// Four consecutive K-fragments (frag stride 1024B): 8x global_load_b128
// issued back-to-back, one s_wait_loadcnt. Amortizes L2 latency over 4 WMMAs.
__device__ __forceinline__ void load_b4(const _Float16* p, v16h* f) {
  v8h o0, o1, o2, o3, o4, o5, o6, o7;
  asm volatile(
      "global_load_b128 %0, %8, off\n\t"
      "global_load_b128 %1, %8, off offset:16\n\t"
      "global_load_b128 %2, %8, off offset:1024\n\t"
      "global_load_b128 %3, %8, off offset:1040\n\t"
      "global_load_b128 %4, %8, off offset:2048\n\t"
      "global_load_b128 %5, %8, off offset:2064\n\t"
      "global_load_b128 %6, %8, off offset:3072\n\t"
      "global_load_b128 %7, %8, off offset:3088\n\t"
      "s_wait_loadcnt 0x0"
      : "=v"(o0), "=v"(o1), "=v"(o2), "=v"(o3),
        "=v"(o4), "=v"(o5), "=v"(o6), "=v"(o7)
      : "v"(p));
  f[0] = mkfrag(o0, o1);
  f[1] = mkfrag(o2, o3);
  f[2] = mkfrag(o4, o5);
  f[3] = mkfrag(o6, o7);
}

// ---------------------------------------------------------------------------
// Weight pack kernel: W[Nout][Kin] (row-major f32, computing s @ W^T) ->
// B-matrix fragments for v_wmma_f32_16x16x32_f16.
// Fragment (kt,nt) is 32x16 f16 = 1024B at frag*1024 (frag = nt*KT+kt);
// within it lane l owns 32 contiguous bytes: B[k = (i<8 ? 8*(l/16) + i :
// 16 + 8*(l/16) + i-8)][n = nt*16 + l%16], i=0..15 (ISA 7.12.2 B layout). Out-of-range k/n padded with 0.
// ---------------------------------------------------------------------------
__global__ void snn_pack_weights(const float* __restrict__ W,
                                 _Float16* __restrict__ dst,
                                 int Kin, int Nout, int KT, int NT) {
  int g    = blockIdx.x * blockDim.x + threadIdx.x;
  int frag = g >> 5;
  int lane = g & 31;
  if (frag >= KT * NT) return;
  int nt  = frag / KT;
  int kt  = frag % KT;
  int hh  = lane >> 4;
  int n   = nt * 16 + (lane & 15);
  _Float16* o = dst + (size_t)frag * 512 + (size_t)lane * 16;
#pragma unroll
  for (int i = 0; i < 16; ++i) {
    int k = kt * 32 + (i < 8 ? hh * 8 + i : 16 + hh * 8 + (i - 8));
    float v = (k < Kin && n < Nout) ? W[(size_t)n * Kin + k] : 0.0f;
    o[i] = (_Float16)v;
  }
}

// ---------------------------------------------------------------------------
// K=256 GEMM tile: z[16x16 f32] = A(LDS row-major f16, stride HID) x B.
// A fragment (ISA 7.12.2): lane l -> row m=l%16; VGPRs 0-3 = K 8h..8h+7,
// VGPRs 4-7 = K 16+8h..  ->  two 16B ds loads per kt.
// ---------------------------------------------------------------------------
__device__ __forceinline__ v8f gemm_tile_k256(const _Float16* S,
                                              const _Float16* __restrict__ pB,
                                              int nt, int lane) {
  const int m  = lane & 15;
  const int hh = lane >> 4;
  const _Float16* base = pB + (size_t)(nt * 8) * 512 + (size_t)lane * 16;
  v8f z = {};
#pragma unroll
  for (int g = 0; g < 2; ++g) {
    v16h bf[4];
    load_b4(base + (size_t)g * 4 * 512, bf);
#pragma unroll
    for (int k = 0; k < 4; ++k) {
      const int kt = g * 4 + k;
      Frag16 a;
      a.h[0] = *(const v8h*)(S + (size_t)m * HID + kt * 32 + 8 * hh);
      a.h[1] = *(const v8h*)(S + (size_t)m * HID + kt * 32 + 16 + 8 * hh);
      z = wmma_f16f32(a.v, bf[k], z);
    }
  }
  return z;
}

// K=32 (layer 1, D_IN padded to 32): single fragment.
__device__ __forceinline__ v8f gemm_tile_k32(const _Float16* S,
                                             const _Float16* __restrict__ pB,
                                             int nt, int lane) {
  const int m  = lane & 15;
  const int hh = lane >> 4;
  const v16h b = load_b1(pB + (size_t)nt * 512 + (size_t)lane * 16);
  Frag16 a;
  a.h[0] = *(const v8h*)(S + (size_t)m * 32 + 8 * hh);
  a.h[1] = *(const v8h*)(S + (size_t)m * 32 + 16 + 8 * hh);
  v8f z = {};
  return wmma_f16f32(a.v, b, z);
}

// Leaky integrate + fire; membrane kept in VGPRs (C/D layout: lane holds
// column n, rows r+8h). Spike (0/1) written as f16 to the next layer's LDS.
__device__ __forceinline__ void leaky_spike(v8f z, float bias, v8f& mem,
                                            _Float16* Sout, int nt, int lane) {
  const int hh = lane >> 4;
  const int n  = nt * 16 + (lane & 15);
#pragma unroll
  for (int r = 0; r < 8; ++r) {
    float mold = mem[r];
    float rst  = (mold > THRF) ? THRF : 0.0f;   // detached prev-spike reset
    float mn   = BETAF * mold + (z[r] + bias) - rst;
    mem[r]     = mn;
    Sout[(size_t)(r + 8 * hh) * HID + n] = (_Float16)((mn > THRF) ? 1.0f : 0.0f);
  }
}

// ---------------------------------------------------------------------------
// Main persistent-tile kernel: block = 256 threads (8 wave32), 16 batch rows.
// ---------------------------------------------------------------------------
__global__ __launch_bounds__(256) void snn_forward(
    const float* __restrict__ x,
    const _Float16* __restrict__ B1, const _Float16* __restrict__ B2,
    const _Float16* __restrict__ B3, const _Float16* __restrict__ B4,
    const _Float16* __restrict__ Bo,
    const float* __restrict__ b1, const float* __restrict__ b2,
    const float* __restrict__ b3, const float* __restrict__ b4,
    float* __restrict__ out) {
  __shared__ __align__(32) _Float16 sX[16][32];          // layer-1 A (K padded)
  __shared__ __align__(32) _Float16 sSp[2][16][HID];     // spike ping-pong
  __shared__ __align__(16) float    sMO[16][208];        // output membranes
  __shared__ float                  sStat[16][2];        // {rowmax, 1/rowsum}

  const int tid     = threadIdx.x;
  const int wave    = tid >> 5;
  const int lane    = tid & 31;
  const int hh      = lane >> 4;
  const int n16     = lane & 15;
  const int rowBase = blockIdx.x * 16;

  // Zero sX once: cols DIN..31 stay zero (K padding for layer 1).
  for (int i = tid; i < 16 * 32; i += 256) ((_Float16*)sX)[i] = (_Float16)0.0f;

  // Per-wave biases for the two owned n-tiles (nt = wave, wave+8).
  float bv1[2], bv2[2], bv3[2], bv4[2];
#pragma unroll
  for (int j = 0; j < 2; ++j) {
    int n = (wave + 8 * j) * 16 + n16;
    bv1[j] = b1[n]; bv2[j] = b2[n]; bv3[j] = b3[n]; bv4[j] = b4[n];
  }

  v8f m1[2] = {}, m2[2] = {}, m3[2] = {}, m4[2] = {};
  v8f mo[2] = {}, acc[2] = {};

  for (int t = 0; t < TSTEPS; ++t) {
    // ---- stage x[:,t,:] into LDS (f32 -> f16) ----
    for (int i = tid; i < 16 * DIN; i += 256) {
      int r = i / DIN, c = i % DIN;
      sX[r][c] = (_Float16)x[((size_t)(rowBase + r) * TSTEPS + t) * DIN + c];
    }
    __syncthreads();

    // ---- layer 1: [16x32] x [32x256] -> spikes in sSp[0] ----
#pragma unroll
    for (int j = 0; j < 2; ++j) {
      const int nt = wave + 8 * j;
      v8f z = gemm_tile_k32(&sX[0][0], B1, nt, lane);
      leaky_spike(z, bv1[j], m1[j], &sSp[0][0][0], nt, lane);
    }
    __syncthreads();

    // ---- layer 2: sSp[0] -> sSp[1] ----
#pragma unroll
    for (int j = 0; j < 2; ++j) {
      const int nt = wave + 8 * j;
      v8f z = gemm_tile_k256(&sSp[0][0][0], B2, nt, lane);
      leaky_spike(z, bv2[j], m2[j], &sSp[1][0][0], nt, lane);
    }
    __syncthreads();

    // ---- layer 3: sSp[1] -> sSp[0] ----
#pragma unroll
    for (int j = 0; j < 2; ++j) {
      const int nt = wave + 8 * j;
      v8f z = gemm_tile_k256(&sSp[1][0][0], B3, nt, lane);
      leaky_spike(z, bv3[j], m3[j], &sSp[0][0][0], nt, lane);
    }
    __syncthreads();

    // ---- layer 4: sSp[0] -> sSp[1] ----
#pragma unroll
    for (int j = 0; j < 2; ++j) {
      const int nt = wave + 8 * j;
      v8f z = gemm_tile_k256(&sSp[0][0][0], B4, nt, lane);
      leaky_spike(z, bv4[j], m4[j], &sSp[1][0][0], nt, lane);
    }
    __syncthreads();

    // ---- output layer: sSp[1] x Wout^T -> mo (no bias, spike unused) ----
#pragma unroll
    for (int j = 0; j < 2; ++j) {
      const int ot = wave + 8 * j;
      if (ot < 13) {
        v8f z = gemm_tile_k256(&sSp[1][0][0], Bo, ot, lane);
#pragma unroll
        for (int r = 0; r < 8; ++r) {
          float mold = mo[j][r];
          float rst  = (mold > THRF) ? THRF : 0.0f;
          float mn   = BETAF * mold + z[r] - rst;
          mo[j][r]   = mn;
          sMO[r + 8 * hh][ot * 16 + n16] = mn;
        }
      }
    }
    __syncthreads();

    // ---- softmax stats per row (only needed when t > warmup) ----
    if (t > WARMUP) {
      const int row = 2 * wave + hh;  // each 16-lane half owns one row
      float vals[13];
      float mx = -3.402823e38f;
#pragma unroll
      for (int jj = 0; jj < 13; ++jj) {
        int c    = n16 + 16 * jj;
        float v  = (c < NOUT) ? sMO[row][c] : -3.402823e38f;
        vals[jj] = v;
        mx       = fmaxf(mx, v);
      }
#pragma unroll
      for (int d = 8; d >= 1; d >>= 1) mx = fmaxf(mx, __shfl_xor(mx, d, 32));
      float sum = 0.0f;
#pragma unroll
      for (int jj = 0; jj < 13; ++jj) {
        int c = n16 + 16 * jj;
        if (c < NOUT) sum += __expf(vals[jj] - mx);
      }
#pragma unroll
      for (int d = 8; d >= 1; d >>= 1) sum += __shfl_xor(sum, d, 32);
      if (n16 == 0) {
        sStat[row][0] = mx;
        sStat[row][1] = 1.0f / sum;
      }
    }
    __syncthreads();

    // ---- acc += softmax(mo) in-register ----
    if (t > WARMUP) {
#pragma unroll
      for (int j = 0; j < 2; ++j) {
        const int ot = wave + 8 * j;
        if (ot < 13) {
          const int c = ot * 16 + n16;
          if (c < NOUT) {
#pragma unroll
            for (int r = 0; r < 8; ++r) {
              float mxr  = sStat[r + 8 * hh][0];
              float invr = sStat[r + 8 * hh][1];
              acc[j][r] += __expf(mo[j][r] - mxr) * invr;
            }
          }
        }
      }
    }
    __syncthreads();
  }

  // ---- write acc [2048 x 200] ----
#pragma unroll
  for (int j = 0; j < 2; ++j) {
    const int ot = wave + 8 * j;
    if (ot < 13) {
      const int c = ot * 16 + n16;
      if (c < NOUT) {
#pragma unroll
        for (int r = 0; r < 8; ++r) {
          out[(size_t)(rowBase + r + 8 * hh) * NOUT + c] = acc[j][r];
        }
      }
    }
  }
}

// ---------------------------------------------------------------------------
// Host-side launcher.
// d_in order: x, W1, b1, W2, b2, W3, b3, W4, b4, Wout
// Workspace layout (f16 elements): B1 @ 0 (16 frags), B2 @ 8192 (128),
// B3 @ 73728, B4 @ 139264, Bo @ 204800 (104 frags). Total 258048 f16 = 504KB.
// ---------------------------------------------------------------------------
extern "C" void kernel_launch(void* const* d_in, const int* in_sizes, int n_in,
                              void* d_out, int out_size, void* d_ws,
                              size_t ws_size, hipStream_t stream) {
  const float* x    = (const float*)d_in[0];
  const float* W1   = (const float*)d_in[1];
  const float* b1   = (const float*)d_in[2];
  const float* W2   = (const float*)d_in[3];
  const float* b2   = (const float*)d_in[4];
  const float* W3   = (const float*)d_in[5];
  const float* b3   = (const float*)d_in[6];
  const float* W4   = (const float*)d_in[7];
  const float* b4   = (const float*)d_in[8];
  const float* Wout = (const float*)d_in[9];

  _Float16* ws  = (_Float16*)d_ws;
  _Float16* B1p = ws;               // KT=1,  NT=16 ->  16 frags
  _Float16* B2p = ws + 8192;        // KT=8,  NT=16 -> 128 frags
  _Float16* B3p = ws + 73728;
  _Float16* B4p = ws + 139264;
  _Float16* Bop = ws + 204800;      // KT=8,  NT=13 -> 104 frags

  // Pack weights (deterministic, cheap; runs every call).
  auto blocks = [](int frags) { return (frags * 32 + 255) / 256; };
  snn_pack_weights<<<blocks(16),  256, 0, stream>>>(W1,   B1p, DIN, HID, 1, 16);
  snn_pack_weights<<<blocks(128), 256, 0, stream>>>(W2,   B2p, HID, HID, 8, 16);
  snn_pack_weights<<<blocks(128), 256, 0, stream>>>(W3,   B3p, HID, HID, 8, 16);
  snn_pack_weights<<<blocks(128), 256, 0, stream>>>(W4,   B4p, HID, HID, 8, 16);
  snn_pack_weights<<<blocks(104), 256, 0, stream>>>(Wout, Bop, HID, NOUT, 8, 13);

  // 128 blocks x 256 threads: each block owns 16 batch rows for all 150 steps.
  snn_forward<<<BATCH / 16, 256, 0, stream>>>(x, B1p, B2p, B3p, B4p, Bop,
                                              b1, b2, b3, b4, (float*)d_out);
}
